// DeeperGCNLayer_87393994539135
// MI455X (gfx1250) — matrix operand order repaired and checked by reading, not compiled, not run on hardware
//
#include <hip/hip_runtime.h>

#define HDIM 128
#define HDIM2 256

typedef __attribute__((ext_vector_type(16))) _Float16 v16h;
typedef __attribute__((ext_vector_type(8)))  float    v8f;

// ---------------------------------------------------------------------------
// Device helpers
// ---------------------------------------------------------------------------
__device__ __forceinline__ void atomic_max_f32(float* addr, float val) {
  // positive floats order like ints, negative floats reverse-order like uints
  if (val >= 0.0f) atomicMax((int*)addr, __float_as_int(val));
  else             atomicMin((unsigned int*)addr, __float_as_uint(val));
}

__device__ __forceinline__ void atomic_add_f32(float* addr, float val) {
  // native CDNA5 HW float atomic, no-return form (STOREcnt)
  asm volatile("global_atomic_add_f32 %0, %1, off scope:SCOPE_DEV"
               :: "v"(addr), "v"(val) : "memory");
}

// ---------------------------------------------------------------------------
// Kernel 1: hn = relu(BN_eval(h))          [N*H elems, float4 vectorized]
// ---------------------------------------------------------------------------
__global__ void bnrelu_kernel(const float* __restrict__ h,
                              const float* __restrict__ bw,
                              const float* __restrict__ bb,
                              const float* __restrict__ bm,
                              const float* __restrict__ bv,
                              float* __restrict__ hn, int NH) {
  int i4 = blockIdx.x * blockDim.x + threadIdx.x;
  int i = i4 * 4;
  if (i >= NH) return;
  int c = i & (HDIM - 1);
  float4 hv = *(const float4*)(h + i);
  float4 wv = *(const float4*)(bw + c);
  float4 biv = *(const float4*)(bb + c);
  float4 mv = *(const float4*)(bm + c);
  float4 vv = *(const float4*)(bv + c);
  float4 o;
  o.x = fmaxf((hv.x - mv.x) * rsqrtf(vv.x + 1e-5f) * wv.x + biv.x, 0.0f);
  o.y = fmaxf((hv.y - mv.y) * rsqrtf(vv.y + 1e-5f) * wv.y + biv.y, 0.0f);
  o.z = fmaxf((hv.z - mv.z) * rsqrtf(vv.z + 1e-5f) * wv.z + biv.z, 0.0f);
  o.w = fmaxf((hv.w - mv.w) * rsqrtf(vv.w + 1e-5f) * wv.w + biv.w, 0.0f);
  *(float4*)(hn + i) = o;
}

// ---------------------------------------------------------------------------
// Kernel 2: init segmax=-inf, num=den=0; build f16 transposed weights
// ---------------------------------------------------------------------------
__global__ void init_ws_kernel(const float* __restrict__ W1,
                               const float* __restrict__ W2,
                               float* __restrict__ segm,
                               float* __restrict__ num,
                               float* __restrict__ den,
                               _Float16* __restrict__ W1T,
                               _Float16* __restrict__ W2T, int NH) {
  int i = blockIdx.x * blockDim.x + threadIdx.x;
  if (i < NH) {
    segm[i] = __uint_as_float(0xff800000u);  // -inf
    num[i] = 0.0f;
    den[i] = 0.0f;
  }
  if (i < HDIM * HDIM2) {
    // W1T[n][k] = W1[k][n], W1 is [128][256] row-major
    int n1 = i >> 7, k1 = i & (HDIM - 1);
    W1T[i] = (_Float16)W1[k1 * HDIM2 + n1];
    // W2T[n][k] = W2[k][n], W2 is [256][128] row-major
    int n2 = i >> 8, k2 = i & (HDIM2 - 1);
    W2T[i] = (_Float16)W2[k2 * HDIM + n2];
  }
}

// ---------------------------------------------------------------------------
// Kernel 3: edge pass 1 — per-dst channelwise max of logits (32 lanes/edge)
// ---------------------------------------------------------------------------
__global__ void edge_max_kernel(const int* __restrict__ ei,
                                const float* __restrict__ ea,
                                const float* __restrict__ hn,
                                const float* __restrict__ tp,
                                float* __restrict__ segm, int E) {
  int gid = blockIdx.x * blockDim.x + threadIdx.x;
  int e = gid >> 5;
  if (e >= E) return;
  int c = (gid & 31) << 2;
  int s = ei[e], d = ei[E + e];
  float t = *tp;
  float4 a = *(const float4*)(ea + (size_t)e * HDIM + c);
  float4 x = *(const float4*)(hn + (size_t)s * HDIM + c);
  float m0 = fmaxf(x.x + a.x, 0.0f) + 1e-7f;
  float m1 = fmaxf(x.y + a.y, 0.0f) + 1e-7f;
  float m2 = fmaxf(x.z + a.z, 0.0f) + 1e-7f;
  float m3 = fmaxf(x.w + a.w, 0.0f) + 1e-7f;
  float* p = segm + (size_t)d * HDIM + c;
  atomic_max_f32(p + 0, m0 * t);
  atomic_max_f32(p + 1, m1 * t);
  atomic_max_f32(p + 2, m2 * t);
  atomic_max_f32(p + 3, m3 * t);
}

// ---------------------------------------------------------------------------
// Kernel 4: segmax fixup: where(isfinite(m), m, 0)
// ---------------------------------------------------------------------------
__global__ void segfix_kernel(float* __restrict__ segm, int NH) {
  int i = blockIdx.x * blockDim.x + threadIdx.x;
  if (i >= NH) return;
  float x = segm[i];
  segm[i] = (fabsf(x) <= 3.0e38f) ? x : 0.0f;  // NaN/±inf -> 0
}

// ---------------------------------------------------------------------------
// Kernel 5: edge pass 2 — num += msg*ex, den += ex  (agg = num/den)
// ---------------------------------------------------------------------------
__global__ void edge_sum_kernel(const int* __restrict__ ei,
                                const float* __restrict__ ea,
                                const float* __restrict__ hn,
                                const float* __restrict__ tp,
                                const float* __restrict__ segm,
                                float* __restrict__ num,
                                float* __restrict__ den, int E) {
  int gid = blockIdx.x * blockDim.x + threadIdx.x;
  int e = gid >> 5;
  if (e >= E) return;
  int c = (gid & 31) << 2;
  int s = ei[e], d = ei[E + e];
  float t = *tp;
  float4 a = *(const float4*)(ea + (size_t)e * HDIM + c);
  float4 x = *(const float4*)(hn + (size_t)s * HDIM + c);
  float m0 = fmaxf(x.x + a.x, 0.0f) + 1e-7f;
  float m1 = fmaxf(x.y + a.y, 0.0f) + 1e-7f;
  float m2 = fmaxf(x.z + a.z, 0.0f) + 1e-7f;
  float m3 = fmaxf(x.w + a.w, 0.0f) + 1e-7f;
  size_t base = (size_t)d * HDIM + c;
  float4 sm = *(const float4*)(segm + base);
  float e0 = __expf(m0 * t - sm.x);
  float e1 = __expf(m1 * t - sm.y);
  float e2 = __expf(m2 * t - sm.z);
  float e3 = __expf(m3 * t - sm.w);
  atomic_add_f32(den + base + 0, e0);
  atomic_add_f32(den + base + 1, e1);
  atomic_add_f32(den + base + 2, e2);
  atomic_add_f32(den + base + 3, e3);
  atomic_add_f32(num + base + 0, m0 * e0);
  atomic_add_f32(num + base + 1, m1 * e1);
  atomic_add_f32(num + base + 2, m2 * e2);
  atomic_add_f32(num + base + 3, m3 * e3);
}

// ---------------------------------------------------------------------------
// Kernel 6: WMMA MLP  out = x + MLP(agg + hn)
//   8 waves/block, 16 nodes/wave, 128 nodes/block.
//   GEMM1: [16x128]@[128x256] via v_wmma_f32_16x16x32_f16 (K-chunks of 32)
//   LayerNorm+ReLU fully in registers (shfl_xor reductions)
//   GEMM2: [16x256]@[256x128], residual add, store.
// ---------------------------------------------------------------------------
__global__ __launch_bounds__(256) void mlp_wmma_kernel(
    const float* __restrict__ x,
    const float* __restrict__ hn,
    const float* __restrict__ num_,
    const float* __restrict__ den_,
    const _Float16* __restrict__ W1T,  // [256][128] f16
    const float* __restrict__ b1,
    const float* __restrict__ lnw,
    const float* __restrict__ lnb,
    const _Float16* __restrict__ W2T,  // [128][256] f16
    const float* __restrict__ b2,
    float* __restrict__ out, int N) {
  extern __shared__ char smem[];  // 64KB, overlaid: outS (32KB) then y16S (64KB)
  const int lane = threadIdx.x & 31;
  const int wave = threadIdx.x >> 5;
  _Float16* outS = (_Float16*)smem + wave * (16 * HDIM);
  _Float16* y16S = (_Float16*)smem + wave * (16 * HDIM2);
  const int row0 = blockIdx.x * 128 + wave * 16;

  const int m = lane & 15;              // row within tile (A/M, or N col for B/C)
  const int hi8 = (lane >> 4) << 3;     // C-row offset
  const int hi16 = (lane >> 4) << 4;    // B-frag K sub-offset
  const int mbase = hi8;

  // ---- Stage A: out = num/(den+1e-16) + hn  -> f16 LDS [16][128] ----
  for (int i = lane; i < 512; i += 32) {
    int r = i >> 5;
    int c = (i & 31) << 2;
    int row = row0 + r;
    float o0 = 0.f, o1 = 0.f, o2 = 0.f, o3 = 0.f;
    if (row < N) {
      size_t base = (size_t)row * HDIM + c;
      float4 nu = *(const float4*)(num_ + base);
      float4 de = *(const float4*)(den_ + base);
      float4 hv = *(const float4*)(hn + base);
      o0 = nu.x / (de.x + 1e-16f) + hv.x;
      o1 = nu.y / (de.y + 1e-16f) + hv.y;
      o2 = nu.z / (de.z + 1e-16f) + hv.z;
      o3 = nu.w / (de.w + 1e-16f) + hv.w;
    }
    outS[r * HDIM + c + 0] = (_Float16)o0;
    outS[r * HDIM + c + 1] = (_Float16)o1;
    outS[r * HDIM + c + 2] = (_Float16)o2;
    outS[r * HDIM + c + 3] = (_Float16)o3;
  }
  __syncthreads();

  // ---- A fragments for GEMM1 (16-bit A 16x32 layout) ----
  v16h a1[4];
#pragma unroll
  for (int kc = 0; kc < 4; ++kc) {
#pragma unroll
    for (int vr = 0; vr < 8; ++vr) {
      int k = kc * 32 + hi16 + (vr << 1);
      a1[kc][2 * vr] = outS[m * HDIM + k];
      a1[kc][2 * vr + 1] = outS[m * HDIM + k + 1];
    }
  }
  __syncthreads();  // outS dead; y16S region may now be written

  // ---- GEMM1: y[16][256] kept in VGPRs (16 C-tiles) ----
  v8f acc[16];
#pragma unroll
  for (int nt = 0; nt < 16; ++nt) {
    int col = nt * 16 + m;
    float bias = b1[col];
    v8f cc;
#pragma unroll
    for (int r = 0; r < 8; ++r) cc[r] = bias;
#pragma unroll
    for (int kc = 0; kc < 4; ++kc) {
      v16h bf = *(const v16h*)(W1T + (size_t)col * HDIM + kc * 32 + hi16);
      cc = __builtin_amdgcn_wmma_f32_16x16x32_f16(false, a1[kc], false, bf,
                                                  (short)0, cc, false, false);
    }
    acc[nt] = cc;
  }

  // ---- LayerNorm over 256 cols, rows live in C layout: reg r <-> M=mbase+r ----
  float s[8], ss[8];
#pragma unroll
  for (int r = 0; r < 8; ++r) { s[r] = 0.f; ss[r] = 0.f; }
#pragma unroll
  for (int nt = 0; nt < 16; ++nt)
#pragma unroll
    for (int r = 0; r < 8; ++r) {
      float v = acc[nt][r];
      s[r] += v;
      ss[r] += v * v;
    }
#pragma unroll
  for (int r = 0; r < 8; ++r) {
#pragma unroll
    for (int mask = 1; mask <= 8; mask <<= 1) {
      s[r] += __shfl_xor(s[r], mask, 32);
      ss[r] += __shfl_xor(ss[r], mask, 32);
    }
  }
  float mu[8], rstd[8];
#pragma unroll
  for (int r = 0; r < 8; ++r) {
    mu[r] = s[r] * (1.0f / 256.0f);
    float var = ss[r] * (1.0f / 256.0f) - mu[r] * mu[r];
    rstd[r] = rsqrtf(var + 1e-5f);
  }
#pragma unroll
  for (int nt = 0; nt < 16; ++nt) {
    int col = nt * 16 + m;
    float w = lnw[col], bb = lnb[col];
#pragma unroll
    for (int r = 0; r < 8; ++r) {
      float v = (acc[nt][r] - mu[r]) * rstd[r] * w + bb;
      v = fmaxf(v, 0.0f);
      y16S[(mbase + r) * HDIM2 + col] = (_Float16)v;
    }
  }
  __syncthreads();

  // ---- A fragments for GEMM2 (K=256 -> 8 chunks) ----
  v16h a2[8];
#pragma unroll
  for (int kc = 0; kc < 8; ++kc) {
#pragma unroll
    for (int vr = 0; vr < 8; ++vr) {
      int k = kc * 32 + hi16 + (vr << 1);
      a2[kc][2 * vr] = y16S[m * HDIM2 + k];
      a2[kc][2 * vr + 1] = y16S[m * HDIM2 + k + 1];
    }
  }

  // ---- GEMM2 + residual + store ----
#pragma unroll
  for (int nt = 0; nt < 8; ++nt) {
    int col = nt * 16 + m;
    float bias = b2[col];
    v8f cc;
#pragma unroll
    for (int r = 0; r < 8; ++r) cc[r] = bias;
#pragma unroll
    for (int kc = 0; kc < 8; ++kc) {
      v16h bf = *(const v16h*)(W2T + (size_t)col * HDIM2 + kc * 32 + hi16);
      cc = __builtin_amdgcn_wmma_f32_16x16x32_f16(false, a2[kc], false, bf,
                                                  (short)0, cc, false, false);
    }
#pragma unroll
    for (int r = 0; r < 8; ++r) {
      int row = row0 + mbase + r;
      if (row < N) {
        size_t idx = (size_t)row * HDIM + col;
        out[idx] = x[idx] + cc[r];
      }
    }
  }
}

// ---------------------------------------------------------------------------
// Host launch
// ---------------------------------------------------------------------------
extern "C" void kernel_launch(void* const* d_in, const int* in_sizes, int n_in,
                              void* d_out, int out_size, void* d_ws, size_t ws_size,
                              hipStream_t stream) {
  const float* h = (const float*)d_in[0];
  const int* ei = (const int*)d_in[1];
  const float* ea = (const float*)d_in[2];
  const float* bw = (const float*)d_in[3];
  const float* bb = (const float*)d_in[4];
  const float* bm = (const float*)d_in[5];
  const float* bv = (const float*)d_in[6];
  const float* t = (const float*)d_in[7];
  const float* W1 = (const float*)d_in[8];
  const float* b1 = (const float*)d_in[9];
  const float* lnw = (const float*)d_in[10];
  const float* lnb = (const float*)d_in[11];
  const float* W2 = (const float*)d_in[12];
  const float* b2 = (const float*)d_in[13];
  float* out = (float*)d_out;

  const int N = in_sizes[0] / HDIM;
  const int E = in_sizes[2] / HDIM;
  const int NH = N * HDIM;

  // workspace layout (floats): hn | segmax | num | den | W1T(f16) | W2T(f16)
  float* hn = (float*)d_ws;
  float* segm = hn + NH;
  float* num = segm + NH;
  float* den = num + NH;
  _Float16* W1T = (_Float16*)(den + NH);
  _Float16* W2T = W1T + HDIM * HDIM2;

  const int TB = 256;
  bnrelu_kernel<<<(NH / 4 + TB - 1) / TB, TB, 0, stream>>>(h, bw, bb, bm, bv, hn, NH);
  init_ws_kernel<<<(NH + TB - 1) / TB, TB, 0, stream>>>(W1, W2, segm, num, den, W1T, W2T, NH);

  const int edgeThreads = E * 32;  // 32 lanes per edge (4 channels each)
  edge_max_kernel<<<(edgeThreads + TB - 1) / TB, TB, 0, stream>>>(ei, ea, hn, t, segm, E);
  segfix_kernel<<<(NH + TB - 1) / TB, TB, 0, stream>>>(segm, NH);
  edge_sum_kernel<<<(edgeThreads + TB - 1) / TB, TB, 0, stream>>>(ei, ea, hn, t, segm, num, den, E);

  const int blocks = (N + 127) / 128;
  mlp_wmma_kernel<<<blocks, 256, 65536, stream>>>(h, hn, num, den, W1T, b1, lnw,
                                                  lnb, W2T, b2, out, N);
}
